// ContextVAE_68530498175289
// MI455X (gfx1250) — matrix operand order repaired and checked by reading, not compiled, not run on hardware
//
#include <hip/hip_runtime.h>
#include <hip/hip_bf16.h>

typedef __attribute__((ext_vector_type(16))) _Float16 v16h;
typedef __attribute__((ext_vector_type(8)))  float    v8f;

#define NB_OBJ 5
#define HID    512

// O_IDS table from the reference (pairs then ordered perms), hardcoded.
__constant__ int c_OIDS[5][12] = {
    {0, 1, 2, 3, 10, 11, 12, 13, 14, 18, 22, 26},
    {0, 4, 5, 6, 10, 14, 15, 16, 17, 19, 23, 27},
    {1, 4, 7, 8, 11, 15, 18, 19, 20, 21, 24, 28},
    {2, 5, 7, 9, 12, 16, 20, 22, 23, 24, 25, 29},
    {3, 6, 8, 9, 13, 17, 21, 25, 26, 27, 28, 29},
};

// ---------------- workspace layout (in _Float16 elements) ----------------
// encW0h [512][64]   @ 0        (32768)
// encW1h [512][512]  @ 32768    (262144)
// Wmh    [16][512]   @ 294912   (8192)
// Wvh    [16][512]   @ 303104   (8192)
// decW0h [512][64]   @ 311296   (32768)
// decW1h [512][512]  @ 344064   (262144)
// decW2h [16][512]   @ 606208   (8192)  -> total 614400 halves (1.2 MB)
#define OFF_ENCW0 0
#define OFF_ENCW1 32768
#define OFF_WM    294912
#define OFF_WV    303104
#define OFF_DECW0 311296
#define OFF_DECW1 344064
#define OFF_DECW2 606208
#define WTOTAL    614400

// ------------------------- weight f32 -> f16 prep ------------------------
__global__ void cvae_prep_weights(const float* encW0, const float* encW1,
                                  const float* Wm, const float* Wv,
                                  const float* decW0, const float* decW1,
                                  const float* decW2, _Float16* ws) {
    int i = blockIdx.x * blockDim.x + threadIdx.x;
    if (i >= WTOTAL) return;
    float val = 0.0f;
    if (i < OFF_ENCW1) {                       // encW0h [512][64], K pad 44->64
        int j = i - OFF_ENCW0, n = j >> 6, k = j & 63;
        if (k < 44) val = encW0[n * 44 + k];
    } else if (i < OFF_WM) {                   // encW1h [512][512]
        val = encW1[i - OFF_ENCW1];
    } else if (i < OFF_WV) {                   // Wmh [16][512], N pad 9->16
        int j = i - OFF_WM, n = j >> 9, k = j & 511;
        if (n < 9) val = Wm[n * 512 + k];
    } else if (i < OFF_DECW0) {                // Wvh
        int j = i - OFF_WV, n = j >> 9, k = j & 511;
        if (n < 9) val = Wv[n * 512 + k];
    } else if (i < OFF_DECW1) {                // decW0h [512][64], K pad 41->64
        int j = i - OFF_DECW0, n = j >> 6, k = j & 63;
        if (k < 41) val = decW0[n * 41 + k];
    } else if (i < OFF_DECW2) {                // decW1h [512][512]
        val = decW1[i - OFF_DECW1];
    } else {                                   // decW2h [16][512], N pad 12->16
        int j = i - OFF_DECW2, n = j >> 9, k = j & 511;
        if (n < 12) val = decW2[n * 512 + k];
    }
    ws[i] = (_Float16)val;
}

// ----------------------- zero the recon output region --------------------
__global__ void cvae_zero_recon(float* recon, int n) {
    int i = blockIdx.x * blockDim.x + threadIdx.x;
    if (i < n) recon[i] = 0.0f;
}

// ------------------------- WMMA fragment loaders --------------------------
// A fragment: documented 16-bit 16x32 A layout. Source is an LDS tile,
// row-major [16][ldk] f16; each VGPR = 2 consecutive K -> one b32 LDS load.
__device__ __forceinline__ v16h load_a_lds(const _Float16* tile, int kbase,
                                           int lane, int ldk) {
    int m = lane & 15, hi = lane >> 4;
    union { v16h h; unsigned u[8]; } f;
    const unsigned* p = (const unsigned*)tile;
#pragma unroll
    for (int j = 0; j < 8; ++j) {
        int blk = j >> 2, w = j & 3;
        int k = kbase + blk * 16 + hi * 8 + w * 2;
        f.u[j] = p[(m * ldk + k) >> 1];
    }
    return f.h;
}

// B fragment: 32x16 K-major tile of W^T, i.e. lane n holds 16 K of weight
// row n in the same table order as the A fragment (k = 16*blk + 8*hi + s).
// Source is global f16 weights, row-major [N][K].
__device__ __forceinline__ v16h load_b_w(const _Float16* Wg, int nbase,
                                         int kbase, int K, int lane) {
    int n = nbase + (lane & 15), hi = lane >> 4;
    union { v16h h; unsigned u[8]; } f;
    const unsigned* p = (const unsigned*)Wg;
#pragma unroll
    for (int j = 0; j < 8; ++j) {
        int k = kbase + (j >> 2) * 16 + hi * 8 + (j & 3) * 2;
        f.u[j] = p[(n * K + k) >> 1];
    }
    return f.h;
}

// Dense layer: out[16][512] = act( in[16][K] * W^T + bias ), f16 LDS tiles.
__device__ __forceinline__ void gemm_layer(const _Float16* inT, _Float16* outT,
                                           const _Float16* Wg, const float* bias,
                                           int K, int lane) {
    int nloc = lane & 15, hi = lane >> 4;
    for (int nb = 0; nb < HID; nb += 16) {
        v8f acc = {};
        for (int kb = 0; kb < K; kb += 32) {
            v16h a = load_a_lds(inT, kb, lane, K);
            v16h b = load_b_w(Wg, nb, kb, K, lane);
            acc = __builtin_amdgcn_wmma_f32_16x16x32_f16(
                false, a, false, b, (short)0, acc, false, false);
        }
        float bv = bias[nb + nloc];
#pragma unroll
        for (int v = 0; v < 8; ++v) {
            float r = acc[v] + bv;
            r = fmaxf(r, 0.0f);
            outT[(v + hi * 8) * HID + nb + nloc] = (_Float16)r;
        }
    }
}

// ------------------------------ fused VAE ---------------------------------
// One wave owns 16 consecutive rows of the flattened [5*B] batch.
// Block = 8 waves = 128 rows; B % 128 == 0 so each block is object-uniform.
__global__ void cvae_fused(const float* __restrict__ initial_c,
                           const float* __restrict__ initial_s,
                           const float* __restrict__ current_c,
                           const float* __restrict__ eps,
                           const float* __restrict__ enc_b0,
                           const float* __restrict__ enc_b1,
                           const float* __restrict__ bm,
                           const float* __restrict__ bv,
                           const float* __restrict__ dec_b0,
                           const float* __restrict__ dec_b1,
                           const float* __restrict__ dec_b2,
                           const _Float16* __restrict__ ws,
                           float* __restrict__ outR,   // [B,30]
                           float* __restrict__ outM,   // [5,B,9]
                           float* __restrict__ outLV,  // [5,B,9]
                           float* __restrict__ outZ,   // [5,B,9]
                           int Bv) {
    extern __shared__ _Float16 smem[];
    const int wave = threadIdx.x >> 5;
    const int lane = threadIdx.x & 31;
    const int r0   = blockIdx.x * 128 + wave * 16;     // first global row
    const int o    = (blockIdx.x * 128) / Bv;          // object id (uniform)

    _Float16* X  = smem + wave * 1024;                 // [16][64]
    _Float16* D  = smem + 8192 + wave * 1024;          // [16][64]
    _Float16* H1 = smem + 16384 + wave * 8192;         // [16][512]
    _Float16* H2 = smem + 81920 + wave * 8192;         // [16][512]

    // ---- build encoder input X = [one_hot(5) | s(15) | gi(12) | gc(12) | 0] ----
    for (int idx = lane; idx < 16 * 64; idx += 32) {
        int m = idx >> 6, c = idx & 63;
        int b = r0 + m - o * Bv;
        float val = 0.0f;
        if (c < 5)        val = (c == o) ? 1.0f : 0.0f;
        else if (c < 20)  val = initial_s[b * 15 + (c - 5)];
        else if (c < 32)  val = initial_c[b * 30 + c_OIDS[o][c - 20]];
        else if (c < 44)  val = current_c[b * 30 + c_OIDS[o][c - 32]];
        X[idx] = (_Float16)val;
    }
    // ---- build decoder input skeleton D = [one_hot(5) | z(9)=later | s(15) | gi(12) | 0]
    for (int idx = lane; idx < 16 * 64; idx += 32) {
        int m = idx >> 6, c = idx & 63;
        int b = r0 + m - o * Bv;
        float val = 0.0f;
        if (c < 5)                     val = (c == o) ? 1.0f : 0.0f;
        else if (c >= 14 && c < 29)    val = initial_s[b * 15 + (c - 14)];
        else if (c >= 29 && c < 41)    val = initial_c[b * 30 + c_OIDS[o][c - 29]];
        D[idx] = (_Float16)val;        // cols 5..13 zero for now (z fills them)
    }

    // ---- encoder ----
    gemm_layer(X,  H1, ws + OFF_ENCW0, enc_b0, 64,  lane);
    gemm_layer(H1, H2, ws + OFF_ENCW1, enc_b1, 512, lane);

    // ---- heads: means & log_var (N padded 9->16), reparameterize ----
    {
        v8f accM = {}, accV = {};
        for (int kb = 0; kb < HID; kb += 32) {
            v16h a  = load_a_lds(H2, kb, lane, HID);
            v16h wm = load_b_w(ws + OFF_WM, 0, kb, HID, lane);
            accM = __builtin_amdgcn_wmma_f32_16x16x32_f16(
                false, a, false, wm, (short)0, accM, false, false);
            v16h wv = load_b_w(ws + OFF_WV, 0, kb, HID, lane);
            accV = __builtin_amdgcn_wmma_f32_16x16x32_f16(
                false, a, false, wv, (short)0, accV, false, false);
        }
        int n = lane & 15, hi = lane >> 4;
        if (n < 9) {
            float bmv = bm[n], bvv = bv[n];
#pragma unroll
            for (int v = 0; v < 8; ++v) {
                int m   = v + hi * 8;
                int row = r0 + m;
                float mean = accM[v] + bmv;
                float lv   = accV[v] + bvv;
                float z    = eps[row * 9 + n] * __expf(0.5f * lv) + mean;
                outM[row * 9 + n]  = mean;
                outLV[row * 9 + n] = lv;
                outZ[row * 9 + n]  = z;
                D[m * 64 + 5 + n]  = (_Float16)z;
            }
        }
    }

    // ---- decoder ----
    gemm_layer(D,  H1, ws + OFF_DECW0, dec_b0, 64,  lane);
    gemm_layer(H1, H2, ws + OFF_DECW1, dec_b1, 512, lane);

    // ---- final head (N padded 12->16), sigmoid, atomic scatter-add ----
    {
        v8f acc = {};
        for (int kb = 0; kb < HID; kb += 32) {
            v16h a = load_a_lds(H2, kb, lane, HID);
            v16h b = load_b_w(ws + OFF_DECW2, 0, kb, HID, lane);
            acc = __builtin_amdgcn_wmma_f32_16x16x32_f16(
                false, a, false, b, (short)0, acc, false, false);
        }
        int n = lane & 15, hi = lane >> 4;
        if (n < 12) {
            float bz  = dec_b2[n];
            int   col = c_OIDS[o][n];
#pragma unroll
            for (int v = 0; v < 8; ++v) {
                int m = v + hi * 8;
                int b = r0 + m - o * Bv;
                float s = 1.0f / (1.0f + __expf(-(acc[v] + bz)));
                atomicAdd(&outR[b * 30 + col], s);
            }
        }
    }
}

extern "C" void kernel_launch(void* const* d_in, const int* in_sizes, int n_in,
                              void* d_out, int out_size, void* d_ws, size_t ws_size,
                              hipStream_t stream) {
    const float* initial_c = (const float*)d_in[0];
    const float* initial_s = (const float*)d_in[1];
    const float* current_c = (const float*)d_in[2];
    const float* eps       = (const float*)d_in[3];
    const float* enc_W0    = (const float*)d_in[4];
    const float* enc_b0    = (const float*)d_in[5];
    const float* enc_W1    = (const float*)d_in[6];
    const float* enc_b1    = (const float*)d_in[7];
    const float* Wm        = (const float*)d_in[8];
    const float* bm        = (const float*)d_in[9];
    const float* Wv        = (const float*)d_in[10];
    const float* bvp       = (const float*)d_in[11];
    const float* dec_W0    = (const float*)d_in[12];
    const float* dec_b0    = (const float*)d_in[13];
    const float* dec_W1    = (const float*)d_in[14];
    const float* dec_b1    = (const float*)d_in[15];
    const float* dec_W2    = (const float*)d_in[16];
    const float* dec_b2    = (const float*)d_in[17];

    const int B = in_sizes[0] / 30;          // 32768
    const int ROWS = NB_OBJ * B;             // 163840

    _Float16* ws = (_Float16*)d_ws;
    float* out   = (float*)d_out;
    float* outR  = out;                      // [B,30]
    float* outM  = out + (size_t)B * 30;     // [5,B,9]
    float* outLV = outM + (size_t)ROWS * 9;
    float* outZ  = outLV + (size_t)ROWS * 9;

    // 1) weights -> padded f16 in workspace (L2-resident, reused by all blocks)
    cvae_prep_weights<<<(WTOTAL + 255) / 256, 256, 0, stream>>>(
        enc_W0, enc_W1, Wm, Wv, dec_W0, dec_W1, dec_W2, ws);

    // 2) zero the scatter-add target
    cvae_zero_recon<<<(B * 30 + 255) / 256, 256, 0, stream>>>(outR, B * 30);

    // 3) fused VAE: 128 rows/block (8 waves x 16 rows), 288 KB LDS/block
    const size_t smemBytes = 147456 * sizeof(_Float16);
    cvae_fused<<<ROWS / 128, 256, smemBytes, stream>>>(
        initial_c, initial_s, current_c, eps,
        enc_b0, enc_b1, bm, bvp, dec_b0, dec_b1, dec_b2,
        ws, outR, outM, outLV, outZ, B);
}
